// DistanceModule_13615046328614
// MI455X (gfx1250) — matrix operand order repaired and checked by reading, not compiled, not run on hardware
//
#include <hip/hip_runtime.h>
#include <hip/hip_bf16.h>

typedef __attribute__((ext_vector_type(16))) _Float16 v16h;
typedef __attribute__((ext_vector_type(8)))  float    v8f;
typedef __attribute__((ext_vector_type(4)))  float    v4f;   // true clang vector (nontemporal-compatible)

#define NB    4
#define NN    768
#define ND    128
#define NC    32
#define NROWS (NB * NN)        // 3072

// ---------------------------------------------------------------------------
// Kernel 1: x = LayerNorm(ReLU(h @ W^T + b)) -> workspace [3072, 32] f32
// One wave computes a 16-row x 32-col tile via 8x v_wmma_f32_16x16x32_f16.
// ---------------------------------------------------------------------------
__global__ __launch_bounds__(256) void gemm_relu_ln_kernel(
    const float* __restrict__ h,      // [3072, 128]
    const float* __restrict__ W,      // [32, 128]
    const float* __restrict__ bias,   // [32]
    const float* __restrict__ gamma,  // [32]
    const float* __restrict__ beta,   // [32]
    float* __restrict__ xout)         // [3072, 32]
{
    __shared__ float tile[8][16][32];           // per-wave private slices

    const int wave = threadIdx.x >> 5;
    const int lane = threadIdx.x & 31;
    const int lr   = lane & 15;                 // row/col-in-tile index
    const int hs   = lane >> 4;                 // lane half select
    const int rbase = (blockIdx.x * 8 + wave) * 16;

    const float* hrow = h + (size_t)(rbase + lr) * ND;
    const float* w0   = W + (size_t)lr * ND;          // N-tile 0: n = lr
    const float* w1   = W + (size_t)(lr + 16) * ND;   // N-tile 1: n = lr+16

    const int abase = hs * 8;    // A: K-offset within 16-chunk
    const int koff  = hs * 16;   // LayerNorm/output channel half [koff, koff+16)

    v8f c0 = {};
    v8f c1 = {};

#pragma unroll
    for (int kb = 0; kb < ND; kb += 32) {
        // ---- A fragment: 16x32 f16 (ISA layout: lane holds K = abase+0..7
        //      in VGPR0-3 halves and K = 16+abase+0..7 in VGPR4-7 halves)
        const v4f a0 = *(const v4f*)(hrow + kb + abase);
        const v4f a1 = *(const v4f*)(hrow + kb + abase + 4);
        const v4f a2 = *(const v4f*)(hrow + kb + 16 + abase);
        const v4f a3 = *(const v4f*)(hrow + kb + 16 + abase + 4);
        v16h a;
        a[0]  = (_Float16)a0.x; a[1]  = (_Float16)a0.y;
        a[2]  = (_Float16)a0.z; a[3]  = (_Float16)a0.w;
        a[4]  = (_Float16)a1.x; a[5]  = (_Float16)a1.y;
        a[6]  = (_Float16)a1.z; a[7]  = (_Float16)a1.w;
        a[8]  = (_Float16)a2.x; a[9]  = (_Float16)a2.y;
        a[10] = (_Float16)a2.z; a[11] = (_Float16)a2.w;
        a[12] = (_Float16)a3.x; a[13] = (_Float16)a3.y;
        a[14] = (_Float16)a3.z; a[15] = (_Float16)a3.w;

        // ---- B fragments: 32x16 f16, B[k][n] = W[n][kb + k]
        //      lane holds K = abase+0..7 and K = 16+abase+0..7 for its column n.
        const v4f q0 = *(const v4f*)(w0 + kb + abase);
        const v4f q1 = *(const v4f*)(w0 + kb + abase + 4);
        const v4f q2 = *(const v4f*)(w0 + kb + 16 + abase);
        const v4f q3 = *(const v4f*)(w0 + kb + 16 + abase + 4);
        v16h b0;
        b0[0]  = (_Float16)q0.x; b0[1]  = (_Float16)q0.y;
        b0[2]  = (_Float16)q0.z; b0[3]  = (_Float16)q0.w;
        b0[4]  = (_Float16)q1.x; b0[5]  = (_Float16)q1.y;
        b0[6]  = (_Float16)q1.z; b0[7]  = (_Float16)q1.w;
        b0[8]  = (_Float16)q2.x; b0[9]  = (_Float16)q2.y;
        b0[10] = (_Float16)q2.z; b0[11] = (_Float16)q2.w;
        b0[12] = (_Float16)q3.x; b0[13] = (_Float16)q3.y;
        b0[14] = (_Float16)q3.z; b0[15] = (_Float16)q3.w;

        const v4f r0 = *(const v4f*)(w1 + kb + abase);
        const v4f r1 = *(const v4f*)(w1 + kb + abase + 4);
        const v4f r2 = *(const v4f*)(w1 + kb + 16 + abase);
        const v4f r3 = *(const v4f*)(w1 + kb + 16 + abase + 4);
        v16h b1;
        b1[0]  = (_Float16)r0.x; b1[1]  = (_Float16)r0.y;
        b1[2]  = (_Float16)r0.z; b1[3]  = (_Float16)r0.w;
        b1[4]  = (_Float16)r1.x; b1[5]  = (_Float16)r1.y;
        b1[6]  = (_Float16)r1.z; b1[7]  = (_Float16)r1.w;
        b1[8]  = (_Float16)r2.x; b1[9]  = (_Float16)r2.y;
        b1[10] = (_Float16)r2.z; b1[11] = (_Float16)r2.w;
        b1[12] = (_Float16)r3.x; b1[13] = (_Float16)r3.y;
        b1[14] = (_Float16)r3.z; b1[15] = (_Float16)r3.w;

        c0 = __builtin_amdgcn_wmma_f32_16x16x32_f16(
                 false, a, false, b0, (short)0, c0, false, false);
        c1 = __builtin_amdgcn_wmma_f32_16x16x32_f16(
                 false, a, false, b1, (short)0, c1, false, false);
    }

    // ---- bias + ReLU, scatter D layout (VGPR v: M = v + 8*hs, N = lr) to LDS
    const float bi0 = bias[lr];
    const float bi1 = bias[lr + 16];
#pragma unroll
    for (int v = 0; v < 8; ++v) {
        const int m = v + 8 * hs;
        tile[wave][m][lr]      = fmaxf(c0[v] + bi0, 0.0f);
        tile[wave][m][lr + 16] = fmaxf(c1[v] + bi1, 0.0f);
    }
    // same-wave LDS producer/consumer: DS ops are in-order, no barrier needed

    // ---- LayerNorm over c=32: lane handles row lr, channel half [koff,koff+16)
    float s = 0.0f, q = 0.0f;
    float vals[16];
#pragma unroll
    for (int k = 0; k < 16; ++k) {
        const float t = tile[wave][lr][koff + k];
        vals[k] = t;
        s += t;
        q += t * t;
    }
    s += __shfl_xor(s, 16, 32);   // combine with the lane holding the other half
    q += __shfl_xor(q, 16, 32);
    const float mu   = s * (1.0f / 32.0f);
    const float var  = q * (1.0f / 32.0f) - mu * mu;
    const float rstd = rsqrtf(var + 1e-5f);

    float* orow = xout + (size_t)(rbase + lr) * NC + koff;
#pragma unroll
    for (int k = 0; k < 16; ++k) {
        orow[k] = (vals[k] - mu) * rstd * gamma[koff + k] + beta[koff + k];
    }
}

// ---------------------------------------------------------------------------
// Kernel 2: out[b,i,j,c] = softmax_c( x[b,i,c] * x[b,j,c] )
// One block per (b,i). Lane split: jg = lane/8 (4 j's per wave),
// cg = lane%8 (4 channels each). Softmax = in-lane reduce(4) + 3x shfl_xor.
// Streaming 302 MB output -> non-temporal b128 stores.
// ---------------------------------------------------------------------------
__global__ __launch_bounds__(256) void pair_softmax_kernel(
    const float* __restrict__ x,   // [3072, 32]
    float* __restrict__ out)       // [4, 768, 768, 32]
{
    const int bi   = blockIdx.x;          // b*768 + i
    const int wave = threadIdx.x >> 5;
    const int lane = threadIdx.x & 31;
    const int jg   = lane >> 3;           // 0..3 : j within wave group
    const int cg   = lane & 7;            // 0..7 : channel quad
    const int brow = (bi / NN) * NN;      // first row of this batch

    const v4f xi = *(const v4f*)(x + (size_t)bi * NC + cg * 4);
    float* outRow = out + (size_t)bi * (NN * NC);

    for (int it = 0; it < NN / 32; ++it) {
        const int j = it * 32 + wave * 4 + jg;
        const v4f xj = *(const v4f*)(x + (size_t)(brow + j) * NC + cg * 4);

        const float px = xi.x * xj.x;
        const float py = xi.y * xj.y;
        const float pz = xi.z * xj.z;
        const float pw = xi.w * xj.w;

        float m = fmaxf(fmaxf(px, py), fmaxf(pz, pw));
        m = fmaxf(m, __shfl_xor(m, 1, 32));
        m = fmaxf(m, __shfl_xor(m, 2, 32));
        m = fmaxf(m, __shfl_xor(m, 4, 32));

        const float ex = __expf(px - m);
        const float ey = __expf(py - m);
        const float ez = __expf(pz - m);
        const float ew = __expf(pw - m);

        float s = (ex + ey) + (ez + ew);
        s += __shfl_xor(s, 1, 32);
        s += __shfl_xor(s, 2, 32);
        s += __shfl_xor(s, 4, 32);
        const float r = 1.0f / s;

        v4f o;
        o.x = ex * r;
        o.y = ey * r;
        o.z = ez * r;
        o.w = ew * r;
        __builtin_nontemporal_store(
            o, (v4f*)(outRow + (size_t)j * NC + cg * 4));
    }
}

// ---------------------------------------------------------------------------
extern "C" void kernel_launch(void* const* d_in, const int* in_sizes, int n_in,
                              void* d_out, int out_size, void* d_ws, size_t ws_size,
                              hipStream_t stream)
{
    (void)in_sizes; (void)n_in; (void)out_size; (void)ws_size;

    const float* h     = (const float*)d_in[0];  // [4,768,128]
    const float* W     = (const float*)d_in[1];  // [32,128]
    const float* b_lin = (const float*)d_in[2];  // [32]
    const float* gamma = (const float*)d_in[3];  // [32]
    const float* beta  = (const float*)d_in[4];  // [32]
    float* out = (float*)d_out;                  // [4,768,768,32]
    float* x   = (float*)d_ws;                   // [3072,32] scratch (384 KiB)

    // Kernel 1: 3072 rows / (8 waves * 16 rows) = 24 blocks
    gemm_relu_ln_kernel<<<NROWS / 128, 256, 0, stream>>>(h, W, b_lin, gamma, beta, x);

    // Kernel 2: one block per (b, i)
    pair_softmax_kernel<<<NROWS, 256, 0, stream>>>(x, out);
}
